// TRGTTemporalRelationAttentionBlock_79860621902503
// MI455X (gfx1250) — matrix operand order repaired and checked by reading, not compiled, not run on hardware
//
#include <hip/hip_runtime.h>
#include <hip/hip_bf16.h>
#include <math.h>

#define NN 50000          // N_NODES
#define NE 600000         // N_EDGES
// HID=128, EDIM=64, HEADS=4, HEAD_DIM=32

typedef _Float16 v16h __attribute__((ext_vector_type(16)));
typedef _Float16 h8v  __attribute__((ext_vector_type(8)));
typedef float    v8f  __attribute__((ext_vector_type(8)));

// ---------------- WMMA helpers (CDNA5 16x16x32 f16 -> f32) ----------------

__device__ __forceinline__ v8f wmma16(v16h a, v16h b, v8f c) {
  return __builtin_amdgcn_wmma_f32_16x16x32_f16(false, a, false, b, (short)0, c,
                                                false, false);
}

// A fragment (16x32, M x K) from a row-major f16 tile with row stride `stride`.
// Lane L holds row M=L%16; halves 0..15 = K(k0 + g*16 + i), g = L>>4
// (matches the B packing).
__device__ __forceinline__ v16h ldA(const _Float16* base, int stride, int k0) {
  const int lane = threadIdx.x & 31;
  const _Float16* p = base + (lane & 15) * stride + k0 + ((lane >> 4) << 4);
  h8v lo = *(const h8v*)p;
  h8v hi = *(const h8v*)(p + 8);
  v16h a;
#pragma unroll
  for (int i = 0; i < 8; ++i) { a[i] = lo[i]; a[i + 8] = hi[i]; }
  return a;
}

// B fragment (32x16, K x N) from the pre-packed buffer: lane L holds column
// N=L%16, K = g*16 + h (this code's contiguous order, same as ldA). Packed so
// each lane's 16 halfs are 32 contiguous bytes.
__device__ __forceinline__ v16h ldB(const _Float16* packed, int frag) {
  const int lane = threadIdx.x & 31;
  const _Float16* p = packed + (frag * 32 + lane) * 16;
  h8v lo = *(const h8v*)p;
  h8v hi = *(const h8v*)(p + 8);
  v16h b;
#pragma unroll
  for (int i = 0; i < 8; ++i) { b[i] = lo[i]; b[i + 8] = hi[i]; }
  return b;
}

__device__ __forceinline__ float gelu_exact(float x) {
  return 0.5f * x * (1.0f + erff(x * 0.70710678118654752f));
}

// monotone uint encoding of float for atomicMax-based segment max
__device__ __forceinline__ unsigned enc_ord(float f) {
  unsigned u = __float_as_uint(f);
  return (u >> 31) ? ~u : (u | 0x80000000u);
}
__device__ __forceinline__ float dec_ord(unsigned u) {
  return (u & 0x80000000u) ? __uint_as_float(u ^ 0x80000000u)
                           : __uint_as_float(~u);
}

// ---------------- small utility kernels ----------------

__global__ __launch_bounds__(256) void init_kernel(unsigned* __restrict__ segmax,
                                                   float* __restrict__ denom,
                                                   float* __restrict__ agg) {
  const int t = blockIdx.x * 256 + threadIdx.x;
  if (t < NN * 128) agg[t] = 0.0f;
  if (t < NN * 4) { denom[t] = 0.0f; segmax[t] = 0x007FFFFFu; /* enc(-inf) */ }
}

// LayerNorm over 128-wide rows, output f16 (one wave per row)
__global__ __launch_bounds__(128)
void ln_f16_kernel(const float* __restrict__ in, const float* __restrict__ w,
                   const float* __restrict__ b, _Float16* __restrict__ out,
                   int nrows) {
  const int row = blockIdx.x * 4 + (threadIdx.x >> 5);
  if (row >= nrows) return;
  const int lane = threadIdx.x & 31;
  const float4 v = ((const float4*)(in + row * 128))[lane];
  float s = v.x + v.y + v.z + v.w;
#pragma unroll
  for (int o = 16; o >= 1; o >>= 1) s += __shfl_xor(s, o, 32);
  const float mu = s * (1.0f / 128.0f);
  const float d0 = v.x - mu, d1 = v.y - mu, d2 = v.z - mu, d3 = v.w - mu;
  float q = d0 * d0 + d1 * d1 + d2 * d2 + d3 * d3;
#pragma unroll
  for (int o = 16; o >= 1; o >>= 1) q += __shfl_xor(q, o, 32);
  const float inv = rsqrtf(q * (1.0f / 128.0f) + 1e-5f);
  const float4 wv = ((const float4*)w)[lane];
  const float4 bv = ((const float4*)b)[lane];
  _Float16* o16 = out + row * 128 + lane * 4;
  o16[0] = (_Float16)(d0 * inv * wv.x + bv.x);
  o16[1] = (_Float16)(d1 * inv * wv.y + bv.y);
  o16[2] = (_Float16)(d2 * inv * wv.z + bv.z);
  o16[3] = (_Float16)(d3 * inv * wv.w + bv.w);
}

// Repack a row-major f32 weight [K x N] into f16 B-fragment order.
// dst[((kt*NT+nt)*32 + lane)*16 + h] = W[(kt*32 + (lane>>4)*16 + h)*N + nt*16 + lane%16]
__global__ __launch_bounds__(256)
void pack_b_kernel(const float* __restrict__ W, _Float16* __restrict__ dst,
                   int K, int N) {
  const int NT = (N + 15) >> 4;
  const int KT = K >> 5;
  const int t = blockIdx.x * 256 + threadIdx.x;
  if (t >= KT * NT * 32) return;
  const int lane = t & 31;
  const int frag = t >> 5;
  const int nt = frag % NT;
  const int kt = frag / NT;
  const int col = nt * 16 + (lane & 15);
  const int g = lane >> 4;
  _Float16* o = dst + (frag * 32 + lane) * 16;
#pragma unroll
  for (int h = 0; h < 16; ++h) {
    const int k = kt * 32 + g * 16 + h;
    o[h] = (_Float16)((col < N) ? W[k * N + col] : 0.0f);
  }
}

// ---------------- edge pass: gate / score / message via WMMA ----------------

__global__ __launch_bounds__(128)
void edge_stage1_kernel(const _Float16* __restrict__ hf16,
                        const float* __restrict__ eemb,
                        const int* __restrict__ esrc,
                        const int* __restrict__ edst,
                        const float* __restrict__ tw,
                        const float* __restrict__ mns,
                        const _Float16* __restrict__ Wm1p, const float* __restrict__ bm1,
                        const _Float16* __restrict__ Wm2p, const float* __restrict__ bm2,
                        const _Float16* __restrict__ Wa1p, const float* __restrict__ ba1,
                        const _Float16* __restrict__ Wa2p, const float* __restrict__ ba2,
                        const _Float16* __restrict__ Wg1p, const float* __restrict__ bg1,
                        const _Float16* __restrict__ Wg2p, const float* __restrict__ bg2,
                        float* __restrict__ outEdge,
                        float* __restrict__ scoreW,
                        unsigned* __restrict__ segmax) {
  __shared__ __align__(16) _Float16 ctxS[4][16][320];  // [hd | hs | ee]
  __shared__ __align__(16) _Float16 stgS[4][16][128];  // hidden staging (reused)
  __shared__ float coefS[4][16][4];                    // gate*time*edge_scale

  const int wave = threadIdx.x >> 5;
  const int lane = threadIdx.x & 31;
  const int tile = blockIdx.x * 4 + wave;  // NE/16 tiles, grid sized exactly
  const int e0 = tile * 16;
  _Float16 (*ctx)[320] = ctxS[wave];
  _Float16* stg = &stgS[wave][0][0];
  float (*coef)[4] = coefS[wave];
  const int m16 = lane & 15;
  const int part = lane >> 4;
  const int mbase = part << 3;

  // gather ctx rows into LDS (f16); ee converted f32->f16 on the fly
  {
    const int e = e0 + m16;
    const int s = esrc[e];
    const int d = edst[e];
    const _Float16* hsr = hf16 + s * 128 + part * 64;
    const _Float16* hdr = hf16 + d * 128 + part * 64;
    _Float16* crow = ctx[m16];
#pragma unroll
    for (int i = 0; i < 64; i += 8) {
      *(h8v*)(crow + part * 64 + i)       = *(const h8v*)(hdr + i);
      *(h8v*)(crow + 128 + part * 64 + i) = *(const h8v*)(hsr + i);
    }
    const float* er = eemb + e * 64 + part * 32;
#pragma unroll
    for (int i = 0; i < 32; ++i) crow[256 + part * 32 + i] = (_Float16)er[i];
    if (part == 0) {
      const float esc = 0.5f * (mns[s] + mns[d]);
#pragma unroll
      for (int hh = 0; hh < 4; ++hh) coef[m16][hh] = tw[e * 4 + hh] * esc;
    }
  }
  // all LDS producers/consumers below are this same wave: DS ops stay in order

  // ---- gate branch: g1 = gelu(ctx@Wg1+bg1); gate = sigmoid(g1@Wg2+bg2) ----
  for (int nt = 0; nt < 8; ++nt) {
    v8f acc = {};
#pragma unroll
    for (int kt = 0; kt < 10; ++kt)  // K = 320
      acc = wmma16(ldA(&ctx[0][0], 320, kt * 32), ldB(Wg1p, kt * 8 + nt), acc);
    const int n = nt * 16 + m16;
    const float bias = bg1[n];
#pragma unroll
    for (int r = 0; r < 8; ++r)
      stg[(mbase + r) * 128 + n] = (_Float16)gelu_exact(acc[r] + bias);
  }
  {
    v8f acc = {};
#pragma unroll
    for (int kt = 0; kt < 4; ++kt)   // K = 128, N padded 4->16
      acc = wmma16(ldA(stg, 128, kt * 32), ldB(Wg2p, kt), acc);
    if (m16 < 4) {
      const float bias = bg2[m16];
#pragma unroll
      for (int r = 0; r < 8; ++r) {
        const float g = 1.0f / (1.0f + expf(-(acc[r] + bias)));
        coef[mbase + r][m16] *= g;
      }
    }
  }

  // ---- attention branch: a1 = gelu(ctx@Wa1+ba1); score = a1@Wa2+ba2 ----
  for (int nt = 0; nt < 8; ++nt) {
    v8f acc = {};
#pragma unroll
    for (int kt = 0; kt < 10; ++kt)
      acc = wmma16(ldA(&ctx[0][0], 320, kt * 32), ldB(Wa1p, kt * 8 + nt), acc);
    const int n = nt * 16 + m16;
    const float bias = ba1[n];
#pragma unroll
    for (int r = 0; r < 8; ++r)
      stg[(mbase + r) * 128 + n] = (_Float16)gelu_exact(acc[r] + bias);
  }
  {
    v8f acc = {};
#pragma unroll
    for (int kt = 0; kt < 4; ++kt)
      acc = wmma16(ldA(stg, 128, kt * 32), ldB(Wa2p, kt), acc);
    if (m16 < 4) {
      const float bias = ba2[m16];
#pragma unroll
      for (int r = 0; r < 8; ++r) {
        const int e = e0 + mbase + r;
        const float sc = (acc[r] + bias) * 0.17677669529663689f;  // 1/sqrt(32)
        scoreW[e * 4 + m16] = sc;
        atomicMax(&segmax[edst[e] * 4 + m16], enc_ord(sc));
      }
    }
  }

  // ---- message branch: msg = gelu([hs|ee]@Wm1+bm1)@Wm2+bm2, scaled ----
  for (int nt = 0; nt < 8; ++nt) {
    v8f acc = {};
#pragma unroll
    for (int kt = 0; kt < 6; ++kt)   // K = 192 (ctx cols 128..319 = [hs|ee])
      acc = wmma16(ldA(&ctx[0][128], 320, kt * 32), ldB(Wm1p, kt * 8 + nt), acc);
    const int n = nt * 16 + m16;
    const float bias = bm1[n];
#pragma unroll
    for (int r = 0; r < 8; ++r)
      stg[(mbase + r) * 128 + n] = (_Float16)gelu_exact(acc[r] + bias);
  }
  for (int nt = 0; nt < 8; ++nt) {
    v8f acc = {};
#pragma unroll
    for (int kt = 0; kt < 4; ++kt)
      acc = wmma16(ldA(stg, 128, kt * 32), ldB(Wm2p, kt * 8 + nt), acc);
    const int n = nt * 16 + m16;
    const int head = nt >> 1;  // 16-col tile stays within one 32-wide head
    const float bias = bm2[n];
#pragma unroll
    for (int r = 0; r < 8; ++r) {
      const int m = mbase + r;
      outEdge[(e0 + m) * 128 + n] = (acc[r] + bias) * coef[m][head];
    }
  }
}

// ---------------- segment softmax + scatter ----------------

__global__ __launch_bounds__(256)
void softmax_exp_kernel(float* __restrict__ scoreW,  // in-place -> exp_s
                        const int* __restrict__ edst,
                        const unsigned* __restrict__ segmax,
                        float* __restrict__ denom) {
  const int t = blockIdx.x * 256 + threadIdx.x;
  if (t >= NE * 4) return;
  const int hh = t & 3;
  const int e = t >> 2;
  const int d = edst[e];
  const float ex = expf(scoreW[t] - dec_ord(segmax[d * 4 + hh]));
  scoreW[t] = ex;
  atomicAdd(&denom[d * 4 + hh], ex);
}

__global__ __launch_bounds__(256)
void scatter_agg_kernel(const float* __restrict__ expW,
                        const float* __restrict__ denom,
                        const int* __restrict__ edst,
                        float* __restrict__ outEdge,  // msg in, edge_repr out
                        float* __restrict__ agg) {
  const int t = blockIdx.x * 256 + threadIdx.x;
  if (t >= NE * 128) return;
  const int c = t & 127;
  const int e = t >> 7;
  const int hh = c >> 5;
  const int d = edst[e];
  const float attn = expW[e * 4 + hh] / fmaxf(denom[d * 4 + hh], 1e-12f);
  const float v = outEdge[t] * attn;
  outEdge[t] = v;
  atomicAdd(&agg[d * 128 + c], v);
}

// ---------------- node update + FFN via WMMA ----------------

__global__ __launch_bounds__(128)
void node_update_kernel(const _Float16* __restrict__ hf16,
                        const float* __restrict__ agg,
                        const float* __restrict__ x,
                        const _Float16* __restrict__ Wselfp, const float* __restrict__ bself,
                        const _Float16* __restrict__ Waggp,  const float* __restrict__ bagg,
                        float* __restrict__ outNode) {
  __shared__ __align__(16) _Float16 hS[4][16][128];
  __shared__ __align__(16) _Float16 aS[4][16][128];
  const int wave = threadIdx.x >> 5;
  const int lane = threadIdx.x & 31;
  const int tile = blockIdx.x * 4 + wave;
  if (tile >= NN / 16) return;  // uniform per wave; EXEC full inside
  const int r0 = tile * 16;
  const int m16 = lane & 15, part = lane >> 4, mbase = part << 3;
  {
    const int nd = r0 + m16;
    const _Float16* hr = hf16 + nd * 128 + part * 64;
    _Float16* hw = &hS[wave][m16][part * 64];
#pragma unroll
    for (int i = 0; i < 64; i += 8) *(h8v*)(hw + i) = *(const h8v*)(hr + i);
    const float* ar = agg + nd * 128 + part * 64;
    _Float16* aw = &aS[wave][m16][part * 64];
#pragma unroll
    for (int i = 0; i < 64; ++i) aw[i] = (_Float16)ar[i];
  }
  for (int nt = 0; nt < 8; ++nt) {
    v8f acc = {};
#pragma unroll
    for (int kt = 0; kt < 4; ++kt) {
      acc = wmma16(ldA(&hS[wave][0][0], 128, kt * 32), ldB(Wselfp, kt * 8 + nt), acc);
      acc = wmma16(ldA(&aS[wave][0][0], 128, kt * 32), ldB(Waggp,  kt * 8 + nt), acc);
    }
    const int n = nt * 16 + m16;
    const float b = bself[n] + bagg[n];
#pragma unroll
    for (int r = 0; r < 8; ++r) {
      const int idx = (r0 + mbase + r) * 128 + n;
      outNode[idx] = x[idx] + acc[r] + b;
    }
  }
}

__global__ __launch_bounds__(128)
void ffn_kernel(const _Float16* __restrict__ h2f16,
                const _Float16* __restrict__ Wf1p, const float* __restrict__ bf1,
                const _Float16* __restrict__ Wf2p, const float* __restrict__ bf2,
                float* __restrict__ outNode) {
  __shared__ __align__(16) _Float16 aS[4][16][128];
  __shared__ __align__(16) _Float16 tS[4][16][256];
  const int wave = threadIdx.x >> 5;
  const int lane = threadIdx.x & 31;
  const int tile = blockIdx.x * 4 + wave;
  if (tile >= NN / 16) return;
  const int r0 = tile * 16;
  const int m16 = lane & 15, part = lane >> 4, mbase = part << 3;
  {
    const _Float16* hr = h2f16 + (r0 + m16) * 128 + part * 64;
    _Float16* hw = &aS[wave][m16][part * 64];
#pragma unroll
    for (int i = 0; i < 64; i += 8) *(h8v*)(hw + i) = *(const h8v*)(hr + i);
  }
  for (int nt = 0; nt < 16; ++nt) {  // N = 256
    v8f acc = {};
#pragma unroll
    for (int kt = 0; kt < 4; ++kt)
      acc = wmma16(ldA(&aS[wave][0][0], 128, kt * 32), ldB(Wf1p, kt * 16 + nt), acc);
    const int n = nt * 16 + m16;
    const float bias = bf1[n];
#pragma unroll
    for (int r = 0; r < 8; ++r)
      tS[wave][mbase + r][n] = (_Float16)gelu_exact(acc[r] + bias);
  }
  for (int nt = 0; nt < 8; ++nt) {
    v8f acc = {};
#pragma unroll
    for (int kt = 0; kt < 8; ++kt)   // K = 256
      acc = wmma16(ldA(&tS[wave][0][0], 256, kt * 32), ldB(Wf2p, kt * 8 + nt), acc);
    const int n = nt * 16 + m16;
    const float bias = bf2[n];
#pragma unroll
    for (int r = 0; r < 8; ++r) {
      const int idx = (r0 + mbase + r) * 128 + n;
      outNode[idx] += acc[r] + bias;
    }
  }
}

// ---------------- host launcher ----------------

extern "C" void kernel_launch(void* const* d_in, const int* in_sizes, int n_in,
                              void* d_out, int out_size, void* d_ws, size_t ws_size,
                              hipStream_t stream) {
  (void)in_sizes; (void)n_in; (void)out_size; (void)ws_size;
  const float* x    = (const float*)d_in[0];
  const int* esrc   = (const int*)d_in[1];
  const int* edst   = (const int*)d_in[2];
  const float* eemb = (const float*)d_in[3];
  const float* tw   = (const float*)d_in[4];
  const float* mns  = (const float*)d_in[5];
  const float* ln1w = (const float*)d_in[6];
  const float* ln1b = (const float*)d_in[7];
  const float* ln2w = (const float*)d_in[8];
  const float* ln2b = (const float*)d_in[9];
  const float* Wself= (const float*)d_in[10]; const float* bself= (const float*)d_in[11];
  const float* Wm1  = (const float*)d_in[12]; const float* bm1  = (const float*)d_in[13];
  const float* Wm2  = (const float*)d_in[14]; const float* bm2  = (const float*)d_in[15];
  const float* Wa1  = (const float*)d_in[16]; const float* ba1  = (const float*)d_in[17];
  const float* Wa2  = (const float*)d_in[18]; const float* ba2  = (const float*)d_in[19];
  const float* Wg1  = (const float*)d_in[20]; const float* bg1  = (const float*)d_in[21];
  const float* Wg2  = (const float*)d_in[22]; const float* bg2  = (const float*)d_in[23];
  const float* Wagg = (const float*)d_in[24]; const float* bagg = (const float*)d_in[25];
  const float* Wf1  = (const float*)d_in[26]; const float* bf1  = (const float*)d_in[27];
  const float* Wf2  = (const float*)d_in[28]; const float* bf2  = (const float*)d_in[29];

  float* outNode = (float*)d_out;                       // [NN,128]
  float* outEdge = outNode + (size_t)NN * 128;          // [NE,128] edge_repr

  // workspace layout (256B aligned regions)
  char* ws = (char*)d_ws;
  size_t off = 0;
  auto alloc = [&](size_t bytes) -> char* {
    char* p = ws + off;
    off = (off + bytes + 255) & ~(size_t)255;
    return p;
  };
  _Float16* hf16   = (_Float16*)alloc((size_t)NN * 128 * 2);  // h1, reused as h2
  float* scoreW    = (float*)alloc((size_t)NE * 4 * 4);       // score -> exp_s
  unsigned* segmax = (unsigned*)alloc((size_t)NN * 4 * 4);
  float* denom     = (float*)alloc((size_t)NN * 4 * 4);
  float* aggB      = (float*)alloc((size_t)NN * 128 * 4);
  _Float16* Wm1p   = (_Float16*)alloc((size_t)6 * 8 * 512 * 2);    // 192x128
  _Float16* Wm2p   = (_Float16*)alloc((size_t)4 * 8 * 512 * 2);    // 128x128
  _Float16* Wa1p   = (_Float16*)alloc((size_t)10 * 8 * 512 * 2);   // 320x128
  _Float16* Wa2p   = (_Float16*)alloc((size_t)4 * 1 * 512 * 2);    // 128x4(pad16)
  _Float16* Wg1p   = (_Float16*)alloc((size_t)10 * 8 * 512 * 2);
  _Float16* Wg2p   = (_Float16*)alloc((size_t)4 * 1 * 512 * 2);
  _Float16* Wselfp = (_Float16*)alloc((size_t)4 * 8 * 512 * 2);
  _Float16* Waggp  = (_Float16*)alloc((size_t)4 * 8 * 512 * 2);
  _Float16* Wf1p   = (_Float16*)alloc((size_t)4 * 16 * 512 * 2);   // 128x256
  _Float16* Wf2p   = (_Float16*)alloc((size_t)8 * 8 * 512 * 2);    // 256x128

  // 1) init accumulators
  init_kernel<<<(NN * 128 + 255) / 256, 256, 0, stream>>>(segmax, denom, aggB);
  // 2) h = LN1(x) -> f16
  ln_f16_kernel<<<NN / 4, 128, 0, stream>>>(x, ln1w, ln1b, hf16, NN);
  // 3) pack weights into B-fragment layout
  auto pack = [&](const float* W, _Float16* dst, int K, int N) {
    const int total = (K / 32) * ((N + 15) / 16) * 32;
    pack_b_kernel<<<(total + 255) / 256, 256, 0, stream>>>(W, dst, K, N);
  };
  pack(Wm1, Wm1p, 192, 128);   pack(Wm2, Wm2p, 128, 128);
  pack(Wa1, Wa1p, 320, 128);   pack(Wa2, Wa2p, 128, 4);
  pack(Wg1, Wg1p, 320, 128);   pack(Wg2, Wg2p, 128, 4);
  pack(Wself, Wselfp, 128, 128); pack(Wagg, Waggp, 128, 128);
  pack(Wf1, Wf1p, 128, 256);   pack(Wf2, Wf2p, 256, 128);
  // 4) edge WMMA pass: gate/score/message; msg written into edge_repr region
  edge_stage1_kernel<<<NE / 16 / 4, 128, 0, stream>>>(
      hf16, eemb, esrc, edst, tw, mns,
      Wm1p, bm1, Wm2p, bm2, Wa1p, ba1, Wa2p, ba2, Wg1p, bg1, Wg2p, bg2,
      outEdge, scoreW, segmax);
  // 5) segment softmax numerator + denominator
  softmax_exp_kernel<<<(NE * 4 + 255) / 256, 256, 0, stream>>>(scoreW, edst,
                                                               segmax, denom);
  // 6) attention scale + scatter-add aggregation (finalizes edge_repr output)
  scatter_agg_kernel<<<(NE * 128) / 256, 256, 0, stream>>>(scoreW, denom, edst,
                                                           outEdge, aggB);
  // 7) out = x + h@Wself + agg@Wagg + biases
  node_update_kernel<<<(NN / 16 + 3) / 4, 128, 0, stream>>>(
      hf16, aggB, x, Wselfp, bself, Waggp, bagg, outNode);
  // 8) h2 = LN2(out) -> f16 (reuse hf16 buffer; h1 fully consumed)
  ln_f16_kernel<<<NN / 4, 128, 0, stream>>>(outNode, ln2w, ln2b, hf16, NN);
  // 9) out += gelu(h2@Wf1+bf1)@Wf2+bf2
  ffn_kernel<<<(NN / 16 + 3) / 4, 128, 0, stream>>>(hf16, Wf1p, bf1, Wf2p, bf2,
                                                    outNode);
}
